// DSAttention_6854767805127
// MI455X (gfx1250) — matrix operand order repaired and checked by reading, not compiled, not run on hardware
//
#include <hip/hip_runtime.h>
#include <hip/hip_bf16.h>
#include <hip/hip_fp16.h>
#include <math.h>

typedef __attribute__((ext_vector_type(16))) _Float16 v16h;
typedef __attribute__((ext_vector_type(8)))  _Float16 v8h;
typedef __attribute__((ext_vector_type(8)))  float    v8f;

#define B_ 2
#define L_ 2048
#define S_ 2048
#define H_ 16
#define E_ 64
#define D_ 64

#define QTILE 128   // queries per workgroup (4 waves x 32)
#define KTILE 64    // keys per S-loop iteration (double-buffered)
#define NITER (S_ / KTILE)
#define NTHREADS 128

// Assemble a 16-wide f16 A fragment from LDS: elements 0..7 at base+off,
// elements 8..15 at base+off+16 (matches the 16-bit A-matrix K-interleave).
__device__ __forceinline__ v16h lds_frag(const _Float16* base, int off) {
    v8h lo = *(const v8h*)(base + off);
    v8h hi = *(const v8h*)(base + off + 16);
    v16h r;
#pragma unroll
    for (int j = 0; j < 8; ++j) { r[j] = lo[j]; r[j + 8] = hi[j]; }
    return r;
}

// Flash attention, transposed formulation, 2 query sub-tiles per wave:
//   S^T = K (S x E) * Q^T (E x L)   -> per-wave 16-key x 16-query D tiles
//   O^T = V^T (D x S) * P^T (S x L) -> per-wave 16-d x 16-query D tiles
// softmax(scale*(tau*QK + delta)) = exp2( st'*QK + sd' - m ) / l
//   st' = scale*tau*log2e folded into Q; sd' = scale*log2e*delta folded into WMMA C.
__global__ __launch_bounds__(NTHREADS)
void ds_attention_fa_kernel(const float* __restrict__ q,
                            const float* __restrict__ k,
                            const float* __restrict__ v,
                            const float* __restrict__ tau,
                            const float* __restrict__ delta,
                            float* __restrict__ out)
{
    __shared__ __align__(16) _Float16 ldsK[2][KTILE * E_];   // K tile  [s][e], f16
    __shared__ __align__(16) _Float16 ldsVT[2][D_ * KTILE];  // V tile transposed [d][s]
    __shared__ __align__(16) float    ldsSD[2][KTILE];       // scale*log2e*delta
    __shared__ __align__(16) _Float16 ldsP[8][16 * KTILE];   // per-(wave,qt) P relayout

    const int tid  = threadIdx.x;
    const int wave = tid >> 5;
    const int lane = tid & 31;
    const int n    = lane & 15;   // fragment column (query / key row / d row)
    const int hf   = lane >> 4;   // half-wave

    const int h = blockIdx.y;
    const int b = blockIdx.z;
    const int qbase = blockIdx.x * QTILE + wave * 32;   // 32 queries per wave

    const float LOG2E   = 1.44269504088896340736f;
    const float st      = 0.125f * tau[b] * LOG2E;   // 1/sqrt(64) * tau * log2e
    const float sdScale = 0.125f * LOG2E;

    // staging geometry (128 threads move a 64x64 f32 tile of K and of V)
    //  K: one row x 32 contiguous elements per thread
    const int srow = tid >> 1;          // key row within tile (0..63)
    const int c32  = (tid & 1) * 32;    // element chunk
    //  V: row-pair x 16 d-columns; transposed layout makes pairs adjacent halfs
    const int vs2  = (tid >> 2) * 2;    // key row pair base (0,2,..,62)
    const int vd16 = (tid & 3) * 16;    // d column block

#define STAGE(BUF, SBASE) do {                                                          \
        const float* kp = k + (((size_t)b * S_ + (SBASE) + srow) * H_ + h) * E_ + c32;  \
        _Pragma("unroll")                                                               \
        for (int g4 = 0; g4 < 4; ++g4) {                                                \
            float4 fa = ((const float4*)kp)[2 * g4];                                    \
            float4 fb = ((const float4*)kp)[2 * g4 + 1];                                \
            v8h kd;                                                                     \
            kd[0]=(_Float16)fa.x; kd[1]=(_Float16)fa.y; kd[2]=(_Float16)fa.z; kd[3]=(_Float16)fa.w; \
            kd[4]=(_Float16)fb.x; kd[5]=(_Float16)fb.y; kd[6]=(_Float16)fb.z; kd[7]=(_Float16)fb.w; \
            *(v8h*)&ldsK[BUF][srow * E_ + c32 + g4 * 8] = kd;                           \
        }                                                                               \
        const float* vp0 = v + (((size_t)b * S_ + (SBASE) + vs2) * H_ + h) * D_ + vd16; \
        const float* vp1 = vp0 + (size_t)H_ * D_;            /* next key row */         \
        _Pragma("unroll")                                                               \
        for (int g4 = 0; g4 < 4; ++g4) {                                                \
            float4 fa = ((const float4*)vp0)[g4];                                       \
            float4 fb = ((const float4*)vp1)[g4];                                       \
            *(__half2*)&ldsVT[BUF][(vd16 + g4*4 + 0) * KTILE + vs2] = __floats2half2_rn(fa.x, fb.x); \
            *(__half2*)&ldsVT[BUF][(vd16 + g4*4 + 1) * KTILE + vs2] = __floats2half2_rn(fa.y, fb.y); \
            *(__half2*)&ldsVT[BUF][(vd16 + g4*4 + 2) * KTILE + vs2] = __floats2half2_rn(fa.z, fb.z); \
            *(__half2*)&ldsVT[BUF][(vd16 + g4*4 + 3) * KTILE + vs2] = __floats2half2_rn(fa.w, fb.w); \
        }                                                                               \
        if (tid < KTILE)                                                                \
            ldsSD[BUF][tid] = sdScale * delta[(size_t)b * S_ + (SBASE) + tid];          \
    } while (0)

    // ---- Q fragments: B-matrix of K*Q^T, per query tile qt and E-chunk c ----
    // B layout (32x16 f16): lane (n,hf), element i -> k = hf*8+i (i<8), 16+hf*8+(i-8) (i>=8)
    v16h bq[2][2];
#pragma unroll
    for (int qt = 0; qt < 2; ++qt) {
        const float* qp = q + (((size_t)b * L_ + qbase + qt * 16 + n) * H_ + h) * E_;
#pragma unroll
        for (int c = 0; c < 2; ++c) {
            const float4* p4 = (const float4*)(qp + c * 32 + hf * 8);
#pragma unroll
            for (int j = 0; j < 4; ++j) {
                float4 f = p4[j + (j >> 1) * 2];
                bq[qt][c][j*4+0] = (_Float16)(f.x * st);
                bq[qt][c][j*4+1] = (_Float16)(f.y * st);
                bq[qt][c][j*4+2] = (_Float16)(f.z * st);
                bq[qt][c][j*4+3] = (_Float16)(f.w * st);
            }
        }
    }

    v8f acc[2][4];                                // O^T d-tiles per query tile
#pragma unroll
    for (int qt = 0; qt < 2; ++qt)
#pragma unroll
        for (int dt = 0; dt < 4; ++dt) acc[qt][dt] = (v8f){};
    float m_run[2] = { -INFINITY, -INFINITY };
    float l_run[2] = { 0.0f, 0.0f };

    STAGE(0, 0);                                  // prologue: stage tile 0
    __syncthreads();

    for (int it = 0; it < NITER; ++it) {
        const int buf = it & 1;
        const int s0  = it * KTILE;

        if (it + 1 < NITER) STAGE(buf ^ 1, s0 + KTILE);   // stage ahead (other buffer)
        if (it + 2 < NITER) {                             // prefetch 2 tiles ahead
            __builtin_prefetch(k + (((size_t)b*S_ + s0 + 2*KTILE + srow)*H_ + h)*E_ + c32, 0, 3);
            __builtin_prefetch(v + (((size_t)b*S_ + s0 + 2*KTILE + vs2)*H_ + h)*D_ + vd16, 0, 3);
        }

        v16h bp[2][2];   // P^T B-fragments per query tile / key group

        // ======== per query tile: scores + online softmax (sequential: limits VGPRs) ==
#pragma unroll
        for (int qt = 0; qt < 2; ++qt) {
            // S^T = K * Q^T, delta-shift folded into WMMA C-init
            // D layout: lane (n,hf), element (nt,r) -> key slot = nt*16 + hf*8 + r
            v8f cs[4];
#pragma unroll
            for (int nt = 0; nt < 4; ++nt) {
                float4 sd0 = *(const float4*)&ldsSD[buf][nt * 16 + hf * 8];
                float4 sd1 = *(const float4*)&ldsSD[buf][nt * 16 + hf * 8 + 4];
                cs[nt][0]=sd0.x; cs[nt][1]=sd0.y; cs[nt][2]=sd0.z; cs[nt][3]=sd0.w;
                cs[nt][4]=sd1.x; cs[nt][5]=sd1.y; cs[nt][6]=sd1.z; cs[nt][7]=sd1.w;
            }
#pragma unroll
            for (int nt = 0; nt < 4; ++nt) {
                const _Float16* kb = &ldsK[buf][(nt * 16 + n) * E_];
                v16h a0 = lds_frag(kb, 0  + hf * 8);      // E-chunk 0
                v16h a1 = lds_frag(kb, 32 + hf * 8);      // E-chunk 1
                cs[nt] = __builtin_amdgcn_wmma_f32_16x16x32_f16(false, a0, false, bq[qt][0],
                                                                (short)0, cs[nt], false, false);
                cs[nt] = __builtin_amdgcn_wmma_f32_16x16x32_f16(false, a1, false, bq[qt][1],
                                                                (short)0, cs[nt], false, false);
            }

            // online softmax over 64 keys (per query column n)
            float lm = cs[0][0];
#pragma unroll
            for (int nt = 0; nt < 4; ++nt)
#pragma unroll
                for (int r = 0; r < 8; ++r) lm = fmaxf(lm, cs[nt][r]);
            lm = fmaxf(lm, __shfl_xor(lm, 16, 32));       // combine the two half-waves
            const float m_new = fmaxf(m_run[qt], lm);
            const float corr  = exp2f(m_run[qt] - m_new);

            // packed-f16 tail: cvt -> sub -> exp2 -> pair-sum, stored straight to LDS
            _Float16* myP = &ldsP[wave * 2 + qt][n * KTILE];
            const __half2 m2 = __float2half2_rn(m_new);
            __half2 psum = __float2half2_rn(0.0f);
#pragma unroll
            for (int nt = 0; nt < 4; ++nt) {
                union { __half2 h2[4]; v8h v; } pk;
#pragma unroll
                for (int j = 0; j < 4; ++j) {
                    __half2 xh = __floats2half2_rn(cs[nt][2*j], cs[nt][2*j + 1]);
                    __half2 e  = h2exp2(__hsub2(xh, m2));
                    psum = __hadd2(psum, e);
                    pk.h2[j] = e;
                }
                // slots nt*16 + hf*8 + (0..7): D-fragment order, contiguous in row
                *(v8h*)&myP[nt * 16 + hf * 8] = pk.v;
            }
            float lsum = __low2float(psum) + __high2float(psum);
            lsum += __shfl_xor(lsum, 16, 32);
            l_run[qt] = l_run[qt] * corr + lsum;
            m_run[qt] = m_new;
#pragma unroll
            for (int dt = 0; dt < 4; ++dt)
#pragma unroll
                for (int r = 0; r < 8; ++r) acc[qt][dt][r] *= corr;

            // read P back in P^T B-fragment order (same-wave LDS, in-order)
            // B element i of group g -> key slot g*32 + hf*8 + i (i<8), g*32 + 16 + hf*8 + (i-8) (i>=8)
#pragma unroll
            for (int g = 0; g < 2; ++g) {
                v8h lo = *(const v8h*)&myP[g * 32 + hf * 8];
                v8h hi = *(const v8h*)&myP[g * 32 + hf * 8 + 16];
#pragma unroll
                for (int j = 0; j < 8; ++j) { bp[qt][g][j] = lo[j]; bp[qt][g][j + 8] = hi[j]; }
            }
        }

        // ===== O^T += V^T * P^T : V^T fragments loaded ONCE, used by both tiles =====
#pragma unroll
        for (int dt = 0; dt < 4; ++dt) {
            const _Float16* vb = &ldsVT[buf][(dt * 16 + n) * KTILE];
            v16h a0 = lds_frag(vb, 0  + hf * 8);      // key slots 0..31 of tile
            v16h a1 = lds_frag(vb, 32 + hf * 8);      // key slots 32..63 of tile
#pragma unroll
            for (int qt = 0; qt < 2; ++qt) {
                acc[qt][dt] = __builtin_amdgcn_wmma_f32_16x16x32_f16(false, a0, false, bp[qt][0],
                                                                     (short)0, acc[qt][dt], false, false);
                acc[qt][dt] = __builtin_amdgcn_wmma_f32_16x16x32_f16(false, a1, false, bp[qt][1],
                                                                     (short)0, acc[qt][dt], false, false);
            }
        }

        __syncthreads();   // staged buf^1 visible; buf free for reuse next+1
    }

    // ================= epilogue: out[b, qrow, h, d] = acc / l =================
#pragma unroll
    for (int qt = 0; qt < 2; ++qt) {
        const float inv_l = 1.0f / l_run[qt];
        float* op = out + (((size_t)b * L_ + qbase + qt * 16 + n) * H_ + h) * D_;
#pragma unroll
        for (int dt = 0; dt < 4; ++dt) {
            // element r -> d = dt*16 + hf*8 + r (8 contiguous floats)
            float4 o0, o1;
            o0.x = acc[qt][dt][0] * inv_l; o0.y = acc[qt][dt][1] * inv_l;
            o0.z = acc[qt][dt][2] * inv_l; o0.w = acc[qt][dt][3] * inv_l;
            o1.x = acc[qt][dt][4] * inv_l; o1.y = acc[qt][dt][5] * inv_l;
            o1.z = acc[qt][dt][6] * inv_l; o1.w = acc[qt][dt][7] * inv_l;
            *(float4*)(op + dt * 16 + hf * 8)     = o0;
            *(float4*)(op + dt * 16 + hf * 8 + 4) = o1;
        }
    }
#undef STAGE
}

extern "C" void kernel_launch(void* const* d_in, const int* in_sizes, int n_in,
                              void* d_out, int out_size, void* d_ws, size_t ws_size,
                              hipStream_t stream) {
    (void)in_sizes; (void)n_in; (void)out_size; (void)d_ws; (void)ws_size;
    const float* q     = (const float*)d_in[0];
    const float* k     = (const float*)d_in[1];
    const float* v     = (const float*)d_in[2];
    const float* tau   = (const float*)d_in[3];
    const float* delta = (const float*)d_in[4];
    // d_in[5] = attn_mask: unused (mask_flag=False in reference)
    float* out = (float*)d_out;

    dim3 grid(L_ / QTILE, H_, B_);   // (16, 16, 2) = 512 workgroups
    ds_attention_fa_kernel<<<grid, NTHREADS, 0, stream>>>(q, k, v, tau, delta, out);
}
